// MoE_41729902247942
// MI455X (gfx1250) — compile-verified, matrix operand order checked by reading, not run on hardware
//
#include <hip/hip_runtime.h>

// ---------------------------------------------------------------------------
// MoE top-1 (DeepSpeed-style) for gfx1250.
// S=8192 tokens, M=1024, E=8, H=4096, C=1024.
// Heavy GEMMs run as bf16 WMMA (v_wmma_f32_16x16x32_bf16) with fp32 accum.
// K-tile 64: two WMMA K-steps per LDS stage (half the barriers per wmma).
// ---------------------------------------------------------------------------

#define S_TOK 8192
#define M_DIM 1024
#define E_NUM 8
#define H_DIM 4096
#define C_CAP 1024

typedef __attribute__((ext_vector_type(16))) __bf16 v16bf;
typedef __attribute__((ext_vector_type(8)))  __bf16 v8bf;
typedef __attribute__((ext_vector_type(8)))  float  v8f;

__device__ __forceinline__ unsigned short f2bf(float f) {
    unsigned u = __builtin_bit_cast(unsigned, f);
    u += 0x7fffu + ((u >> 16) & 1u);      // round-to-nearest-even
    return (unsigned short)(u >> 16);
}
__device__ __forceinline__ unsigned pack2bf(float lo, float hi) {
    return (unsigned)f2bf(lo) | ((unsigned)f2bf(hi) << 16);
}

// ---------------------------------------------------------------------------
// 1) Gate: logits = x[s,:] @ wg -> softmax -> (argmax expert, prob)
// ---------------------------------------------------------------------------
__global__ void gate_kernel(const float* __restrict__ x,
                            const float* __restrict__ wg,
                            int* __restrict__ eidx,
                            float* __restrict__ prob) {
    const int s   = blockIdx.x;
    const int tid = threadIdx.x;                 // 128 threads
    const float* xr = x + (size_t)s * M_DIM;

    float p[E_NUM];
#pragma unroll
    for (int e = 0; e < E_NUM; ++e) p[e] = 0.f;

    for (int t = tid; t < M_DIM; t += 128) {
        const float xv = xr[t];
        const float* wr = wg + (size_t)t * E_NUM;
#pragma unroll
        for (int e = 0; e < E_NUM; ++e) p[e] += xv * wr[e];
    }

    __shared__ float red[128][E_NUM];
#pragma unroll
    for (int e = 0; e < E_NUM; ++e) red[tid][e] = p[e];
    __syncthreads();
    for (int off = 64; off > 0; off >>= 1) {
        if (tid < off) {
#pragma unroll
            for (int e = 0; e < E_NUM; ++e) red[tid][e] += red[tid + off][e];
        }
        __syncthreads();
    }

    if (tid == 0) {
        float m = red[0][0];
        int   am = 0;
#pragma unroll
        for (int e = 1; e < E_NUM; ++e) {
            if (red[0][e] > m) { m = red[0][e]; am = e; }   // first-max, like argmax
        }
        float denom = 0.f;
#pragma unroll
        for (int e = 0; e < E_NUM; ++e) denom += __expf(red[0][e] - m);
        eidx[s] = am;
        prob[s] = 1.f / denom;   // softmax value at the argmax
    }
}

// ---------------------------------------------------------------------------
// 2) Sequential capacity scan: exact cumsum order, one wave32.
// ---------------------------------------------------------------------------
__global__ void scan_kernel(const int* __restrict__ eidx,
                            const float* __restrict__ prob,
                            int* __restrict__ slot,
                            float* __restrict__ gateval) {
    const int lane = threadIdx.x;                // 32 threads, one wave
    __shared__ int counters[E_NUM];
    __shared__ int eshare[32];
    if (lane < E_NUM) counters[lane] = 0;
    __syncthreads();

    for (int c = 0; c < S_TOK / 32; ++c) {
        const int s = c * 32 + lane;
        const int e = eidx[s];
        eshare[lane] = e;
        __syncthreads();

        int myprefix = 0, total = 0;
#pragma unroll
        for (int t = 0; t < 32; ++t) {
            if (eshare[t] == e) { total++; if (t < lane) myprefix++; }
        }
        const int base = counters[e];
        __syncthreads();
        if (myprefix == total - 1) counters[e] = base + total;  // last occurrence updates
        const int pos  = base + myprefix;
        const bool keep = (pos < C_CAP);
        slot[s]    = keep ? (e * C_CAP + pos) : -1;
        gateval[s] = keep ? prob[s] : 0.f;
        __syncthreads();
    }
}

// ---------------------------------------------------------------------------
// 3) Zero dispatch buffer (bf16 [E*C, M]); 4) scatter kept tokens (fp32->bf16)
// ---------------------------------------------------------------------------
__global__ void zero_disp_kernel(uint4* __restrict__ disp) {
    const size_t i = (size_t)blockIdx.x * blockDim.x + threadIdx.x;
    uint4 z; z.x = 0u; z.y = 0u; z.z = 0u; z.w = 0u;
    disp[i] = z;                                  // 16 B each, exact grid
}

__global__ void scatter_kernel(const float* __restrict__ x,
                               const int* __restrict__ slot,
                               unsigned short* __restrict__ disp) {
    const int s  = blockIdx.x;
    const int sl = slot[s];
    if (sl < 0) return;
    const int t = threadIdx.x;                    // 256 threads, 4 floats each
    const float4 v = ((const float4*)(x + (size_t)s * M_DIM))[t];
    uint2 pk; pk.x = pack2bf(v.x, v.y); pk.y = pack2bf(v.z, v.w);
    *(uint2*)(disp + (size_t)sl * M_DIM + t * 4) = pk;
}

// ---------------------------------------------------------------------------
// 5/6) Grouped GEMM:  Out[e] = act( A[e](1024 x K, bf16) @ B[e](K x N, fp32->bf16) + bias[e] )
// Workgroup tile 128x128, 8 waves, each wave 64x32 (4x2 WMMA fragments).
// K-tile 64 (two 16x16x32 K-steps per LDS stage). lda == K, ldb == ldo == N.
// ---------------------------------------------------------------------------
template <bool RELU, bool OUT_BF16>
__global__ void moe_gemm(const unsigned short* __restrict__ A,
                         const float* __restrict__ B,
                         const float* __restrict__ bias,
                         void* __restrict__ Out,
                         int N, int K) {
    const int e    = blockIdx.z;
    const int row0 = blockIdx.y * 128;
    const int n0   = blockIdx.x * 128;

    const unsigned short* Ae = A    + (size_t)e * C_CAP * K;
    const float*          Be = B    + (size_t)e * K * N;
    const float*          be = bias + (size_t)e * N;

    const int tid  = threadIdx.x;        // 256
    const int wave = tid >> 5;           // 0..7
    const int lane = tid & 31;
    const int wr   = wave & 1;           // wave row  (64-row band)
    const int wc   = wave >> 1;          // wave col  (32-col band)

    __shared__ __align__(16) unsigned short Atile[128][64];  // [row][k]   bf16 (16 KB)
    __shared__ __align__(16) unsigned short Btile[128][64];  // [n][k]     bf16 (16 KB, transposed)

    v8f acc[4][2];
    const v8f vzero = {0.f, 0.f, 0.f, 0.f, 0.f, 0.f, 0.f, 0.f};
#pragma unroll
    for (int i = 0; i < 4; ++i)
#pragma unroll
        for (int j = 0; j < 2; ++j) acc[i][j] = vzero;

    const int r    = lane & 15;
    const int koff = (lane >> 4) << 3;   // 0 or 8 : ISA 16-bit A/B register layout

    for (int k0 = 0; k0 < K; k0 += 64) {
        // --- A tile fill: 128x64 bf16 (16 KB), 16 B per thread x4 ---
#pragma unroll
        for (int i = 0; i < 4; ++i) {
            const int q   = i * 256 + tid;       // 0..1023 chunks of 8 bf16
            const int row = q >> 3;
            const int ch  = q & 7;
            const uint4 v = *(const uint4*)(Ae + (size_t)(row0 + row) * K + k0 + ch * 8);
            *(uint4*)&Atile[row][ch * 8] = v;
        }
        // --- B tile fill: 64 x 128 fp32 -> bf16, transposed [n][k] with
        //     (k,k+1) pairs packed into one b32 store ---
#pragma unroll
        for (int i = 0; i < 4; ++i) {
            const int q  = i * 256 + tid;        // 0..1023 (kpair, float4-col)
            const int kp = q >> 5;               // 0..31 -> rows 2kp, 2kp+1
            const int c4 = q & 31;               // float4 index along n
            const float* rp = Be + (size_t)(k0 + 2 * kp) * N + n0 + c4 * 4;
            const float4 v0 = *(const float4*)(rp);
            const float4 v1 = *(const float4*)(rp + N);
            const int n = c4 * 4;
            *(unsigned*)&Btile[n + 0][2 * kp] = pack2bf(v0.x, v1.x);
            *(unsigned*)&Btile[n + 1][2 * kp] = pack2bf(v0.y, v1.y);
            *(unsigned*)&Btile[n + 2][2 * kp] = pack2bf(v0.z, v1.z);
            *(unsigned*)&Btile[n + 3][2 * kp] = pack2bf(v0.w, v1.w);
        }
        if (k0 + 64 < K)
            __builtin_prefetch(Be + (size_t)(k0 + 64) * N + n0, 0, 1);
        __syncthreads();

        // --- two WMMA K-steps per stage ---
        auto ldfrag = [&](const unsigned short* p) -> v16bf {
            const v8bf lo = *(const v8bf*)(p);        // K .. K+7
            const v8bf hi = *(const v8bf*)(p + 16);   // K+16 .. K+23
            return __builtin_shufflevector(lo, hi, 0, 1, 2, 3, 4, 5, 6, 7,
                                                   8, 9, 10, 11, 12, 13, 14, 15);
        };
#pragma unroll
        for (int kk = 0; kk < 64; kk += 32) {
            v16bf af[4], bf[2];
#pragma unroll
            for (int i = 0; i < 4; ++i)
                af[i] = ldfrag(&Atile[wr * 64 + i * 16 + r][kk + koff]);
#pragma unroll
            for (int j = 0; j < 2; ++j)
                bf[j] = ldfrag(&Btile[wc * 32 + j * 16 + r][kk + koff]);

#pragma unroll
            for (int i = 0; i < 4; ++i)
#pragma unroll
                for (int j = 0; j < 2; ++j)
                    acc[i][j] = __builtin_amdgcn_wmma_f32_16x16x32_bf16(
                        false, af[i], false, bf[j], (short)0, acc[i][j], false, false);
        }
        __syncthreads();
    }

    // --- epilogue: bias (+ReLU), store fp32 or bf16 ---
    const int rsel = (lane >> 4) * 8;    // C/D layout: lanes 16-31 hold rows +8
#pragma unroll
    for (int i = 0; i < 4; ++i) {
#pragma unroll
        for (int j = 0; j < 2; ++j) {
            const int col  = n0 + wc * 32 + j * 16 + r;
            const int rowb = row0 + wr * 64 + i * 16 + rsel;
            const float bv = be[col];
#pragma unroll
            for (int jj = 0; jj < 8; ++jj) {
                float v = acc[i][j][jj] + bv;
                if (RELU) v = fmaxf(v, 0.f);
                const size_t off = (size_t)(rowb + jj) * N + col + (size_t)e * C_CAP * N;
                if (OUT_BF16) ((unsigned short*)Out)[off] = f2bf(v);
                else          ((float*)Out)[off]          = v;
            }
        }
    }
}

// ---------------------------------------------------------------------------
// 7) Combine: out[s,:] = gate_val[s] * eo[slot[s],:]  (0 for dropped tokens)
// ---------------------------------------------------------------------------
__global__ void combine_kernel(const float* __restrict__ eo,
                               const int* __restrict__ slot,
                               const float* __restrict__ gateval,
                               float* __restrict__ out) {
    const int s  = blockIdx.x;
    const int t  = threadIdx.x;                  // 256, 4 floats each
    const int sl = slot[s];
    float4* o = (float4*)(out + (size_t)s * M_DIM);
    if (sl < 0) {
        o[t] = make_float4(0.f, 0.f, 0.f, 0.f);
        return;
    }
    const float g  = gateval[s];
    const float4 v = ((const float4*)(eo + (size_t)sl * M_DIM))[t];
    o[t] = make_float4(v.x * g, v.y * g, v.z * g, v.w * g);
}

// ---------------------------------------------------------------------------
extern "C" void kernel_launch(void* const* d_in, const int* in_sizes, int n_in,
                              void* d_out, int out_size, void* d_ws, size_t ws_size,
                              hipStream_t stream) {
    const float* x  = (const float*)d_in[0];   // [4,2048,1024]
    const float* wg = (const float*)d_in[1];   // [1024,8]
    const float* w1 = (const float*)d_in[2];   // [8,1024,4096]
    const float* b1 = (const float*)d_in[3];   // [8,4096]
    const float* w2 = (const float*)d_in[4];   // [8,4096,1024]
    const float* b2 = (const float*)d_in[5];   // [8,1024]
    float* out = (float*)d_out;

    char* ws = (char*)d_ws;
    int*            eidx    = (int*)(ws + 0);                       //  32 KB
    int*            slot    = (int*)(ws + 32768);                   //  32 KB
    float*          prob    = (float*)(ws + 65536);                 //  32 KB
    float*          gateval = (float*)(ws + 98304);                 //  32 KB
    unsigned short* disp    = (unsigned short*)(ws + 131072);       //  16 MB bf16 [8192,1024]
    unsigned short* hbuf    = (unsigned short*)(ws + 16908288);     //  64 MB bf16 [8192,4096]
    float*          eo      = (float*)(ws + 84017152);              //  32 MB fp32 [8192,1024]

    gate_kernel<<<S_TOK, 128, 0, stream>>>(x, wg, eidx, prob);
    scan_kernel<<<1, 32, 0, stream>>>(eidx, prob, slot, gateval);

    // disp = 8192*1024 bf16 = 16 MB = 1,048,576 uint4
    zero_disp_kernel<<<4096, 256, 0, stream>>>((uint4*)disp);
    scatter_kernel<<<S_TOK, 256, 0, stream>>>(x, slot, disp);

    // GEMM1: h = relu(disp @ w1 + b1)   [per expert 1024x1024 @ 1024x4096]
    moe_gemm<true, true><<<dim3(H_DIM / 128, C_CAP / 128, E_NUM), 256, 0, stream>>>(
        disp, w1, b1, (void*)hbuf, H_DIM, M_DIM);
    // GEMM2: eo = h @ w2 + b2           [per expert 1024x4096 @ 4096x1024]
    moe_gemm<false, false><<<dim3(M_DIM / 128, C_CAP / 128, E_NUM), 256, 0, stream>>>(
        hbuf, w2, b2, (void*)eo, M_DIM, H_DIM);

    combine_kernel<<<S_TOK, 256, 0, stream>>>(eo, slot, gateval, out);
}
